// MSABlock_85581518340274
// MI455X (gfx1250) — compile-verified, matrix operand order checked by reading, not run on hardware
//
#include <hip/hip_runtime.h>
#include <math.h>

// ---------------------------------------------------------------------------
// MSA block for MI455X (gfx1250, wave32, WMMA).
// All heavy contractions (qkv / attention / merge / proj / fc1 / deform-conv-
// as-im2col-GEMM / fc2) run through v_wmma_f32_16x16x32_bf16 with fp32
// accumulation. GEMM waves are 2x4 register-blocked (8 wmma per 12 fragment
// loads, ~21 FLOP/byte); all fragments of a k-step are loaded into distinct
// registers up-front so the loads clause together and the waits stagger
// (no single-buffered B recycling -> no WMMA->VALU WAR v_nops).
// LN / softmax / bilinear gather / offset convs / 8x8 ibpp stay on VALU.
// ---------------------------------------------------------------------------

typedef __attribute__((ext_vector_type(16))) __bf16 v16bf;
typedef __attribute__((ext_vector_type(8)))  __bf16 bf16x8;
typedef __attribute__((ext_vector_type(8)))  float  v8f;

#define BB 2
#define NN 1024
#define CC 256
#define HEADS 8
#define DH 32
#define MM (BB * NN)        // 2048 rows
#define HID 1024
#define H2C 512
#define HH 32
#define WW 32

__device__ __forceinline__ __bf16 f2bf(float f) {
  unsigned u = __float_as_uint(f);
  unsigned r = (u + 0x7FFFu + ((u >> 16) & 1u)) >> 16;   // round-nearest-even
  unsigned short s = (unsigned short)r;
  __bf16 b;
  __builtin_memcpy(&b, &s, 2);
  return b;
}

// 16-bit A/B fragment layout (ISA 7.12.2): lane l holds row/col (l&15);
// kb = (l<16)?0:8. VGPRs 0..3 hold K[kb..kb+8) contiguous, VGPRs 4..7 hold
// K[kb+16..kb+24) contiguous -> two 16-byte loads per fragment.
__device__ __forceinline__ v16bf load_frag(const __bf16* p) {
  bf16x8 lo = *(const bf16x8*)(p);
  bf16x8 hi = *(const bf16x8*)(p + 16);
  v16bf f;
#pragma unroll
  for (int i = 0; i < 8; ++i) { f[i] = lo[i]; f[i + 8] = hi[i]; }
  return f;
}

__device__ __forceinline__ v8f wmma_bf16(v16bf a, v16bf b, v8f c) {
  return __builtin_amdgcn_wmma_f32_16x16x32_bf16(false, a, false, b, (short)0,
                                                 c, false, false);
}

// Epilogue for one 16x16 f32 C tile (ISA C/D layout: VGPR r -> M = r (lanes
// 0-15) / 8+r (lanes 16-31); N = lane&15).
__device__ __forceinline__ void store_tile(v8f acc, int m0, int n0, int lane,
                                           const float* bias,
                                           const float* resid, int ldr,
                                           float* Cf, int ldc, __bf16* Cb,
                                           int ldcb) {
  const int cn = lane & 15;
  const int mb = m0 + ((lane >> 4) << 3);
  const float bia = bias ? bias[n0 + cn] : 0.0f;
#pragma unroll
  for (int r = 0; r < 8; ++r) {
    const int m = mb + r;
    float v = acc[r] + bia;
    if (resid) v += resid[(size_t)m * ldr + n0 + cn];
    if (Cf) Cf[(size_t)m * ldc + n0 + cn] = v;
    if (Cb) Cb[(size_t)m * ldcb + n0 + cn] = f2bf(v);
  }
}

// ---------------------------------------------------------------------------
// Generic bf16 WMMA GEMM:  C[M x N] = A[M x K] * Bt[N x K]^T (+bias)(+resid)
// blockDim = 128 (4 waves, 2x2) -> block tile 64(M) x 128(N).
// Each wave computes a 32x64 tile: 2 A-frags + 4 B-frags, all loaded up-front
// per k-step, then 8 wmma ordered by fragment availability.
// grid = (M/64, N/128); K mult of 32.
// ---------------------------------------------------------------------------
__global__ void __launch_bounds__(128, 1)
gemm_bf16_wmma(const __bf16* __restrict__ A, int lda,
               const __bf16* __restrict__ Bt, int ldb,
               const float* __restrict__ bias,
               const float* __restrict__ resid, int ldr,
               float* __restrict__ Cf, int ldc,
               __bf16* __restrict__ Cb, int ldcb, int K) {
  const int wave = threadIdx.x >> 5;
  const int lane = threadIdx.x & 31;
  const int m0 = blockIdx.x * 64 + (wave >> 1) * 32;
  const int n0 = blockIdx.y * 128 + (wave & 1) * 64;
  const int kb = (lane < 16) ? 0 : 8;
  const __bf16* Ap0 = A + (size_t)(m0 + (lane & 15)) * lda + kb;
  const __bf16* Ap1 = Ap0 + (size_t)16 * lda;
  const __bf16* Bp0 = Bt + (size_t)(n0 + (lane & 15)) * ldb + kb;
  const __bf16* Bp1 = Bp0 + (size_t)16 * ldb;
  const __bf16* Bp2 = Bp0 + (size_t)32 * ldb;
  const __bf16* Bp3 = Bp0 + (size_t)48 * ldb;
  v8f acc00 = {}, acc01 = {}, acc02 = {}, acc03 = {};
  v8f acc10 = {}, acc11 = {}, acc12 = {}, acc13 = {};
  for (int k0 = 0; k0 < K; k0 += 32) {
    v16bf a0 = load_frag(Ap0 + k0);
    v16bf a1 = load_frag(Ap1 + k0);
    v16bf b0 = load_frag(Bp0 + k0);
    v16bf b1 = load_frag(Bp1 + k0);
    v16bf b2 = load_frag(Bp2 + k0);
    v16bf b3 = load_frag(Bp3 + k0);
    acc00 = wmma_bf16(a0, b0, acc00);
    acc10 = wmma_bf16(a1, b0, acc10);
    acc01 = wmma_bf16(a0, b1, acc01);
    acc11 = wmma_bf16(a1, b1, acc11);
    acc02 = wmma_bf16(a0, b2, acc02);
    acc12 = wmma_bf16(a1, b2, acc12);
    acc03 = wmma_bf16(a0, b3, acc03);
    acc13 = wmma_bf16(a1, b3, acc13);
  }
  store_tile(acc00, m0,      n0,      lane, bias, resid, ldr, Cf, ldc, Cb, ldcb);
  store_tile(acc01, m0,      n0 + 16, lane, bias, resid, ldr, Cf, ldc, Cb, ldcb);
  store_tile(acc02, m0,      n0 + 32, lane, bias, resid, ldr, Cf, ldc, Cb, ldcb);
  store_tile(acc03, m0,      n0 + 48, lane, bias, resid, ldr, Cf, ldc, Cb, ldcb);
  store_tile(acc10, m0 + 16, n0,      lane, bias, resid, ldr, Cf, ldc, Cb, ldcb);
  store_tile(acc11, m0 + 16, n0 + 16, lane, bias, resid, ldr, Cf, ldc, Cb, ldcb);
  store_tile(acc12, m0 + 16, n0 + 32, lane, bias, resid, ldr, Cf, ldc, Cb, ldcb);
  store_tile(acc13, m0 + 16, n0 + 48, lane, bias, resid, ldr, Cf, ldc, Cb, ldcb);
}

// ---------------------------------------------------------------------------
// Attention scores: S[bh,i,j] = scale * sum_d Q[i,d] K[j,d]; dh=32 -> single
// k-step. Wave computes 32x64 (2 Q-frags + 4 K-frags up-front, 8 wmma).
// grid (32, 4, B*H), block 128.
// ---------------------------------------------------------------------------
__global__ void __launch_bounds__(128, 1)
attn_scores(const __bf16* __restrict__ qkvb, float* __restrict__ S,
            float scale) {
  const int wave = threadIdx.x >> 5, lane = threadIdx.x & 31;
  const int bh = blockIdx.z, b = bh >> 3, h = bh & 7;
  const int i0 = blockIdx.x * 32;
  const int j0 = (blockIdx.y * 4 + wave) * 64;
  const int kb = (lane < 16) ? 0 : 8;
  const __bf16* Qp0 =
      qkvb + (size_t)(b * NN + i0 + (lane & 15)) * 768 + h * DH + kb;
  const __bf16* Kp =
      qkvb + (size_t)(b * NN + j0 + (lane & 15)) * 768 + CC + h * DH + kb;
  v16bf a0 = load_frag(Qp0);
  v16bf a1 = load_frag(Qp0 + (size_t)16 * 768);
  v16bf b0 = load_frag(Kp);
  v16bf b1 = load_frag(Kp + (size_t)16 * 768);
  v16bf b2 = load_frag(Kp + (size_t)32 * 768);
  v16bf b3 = load_frag(Kp + (size_t)48 * 768);
  v8f acc[2][4] = {};
  acc[0][0] = wmma_bf16(a0, b0, acc[0][0]);
  acc[1][0] = wmma_bf16(a1, b0, acc[1][0]);
  acc[0][1] = wmma_bf16(a0, b1, acc[0][1]);
  acc[1][1] = wmma_bf16(a1, b1, acc[1][1]);
  acc[0][2] = wmma_bf16(a0, b2, acc[0][2]);
  acc[1][2] = wmma_bf16(a1, b2, acc[1][2]);
  acc[0][3] = wmma_bf16(a0, b3, acc[0][3]);
  acc[1][3] = wmma_bf16(a1, b3, acc[1][3]);
  float* Sp = S + (size_t)bh * NN * NN;
  const int cn = lane & 15;
#pragma unroll
  for (int i = 0; i < 2; ++i)
#pragma unroll
    for (int j = 0; j < 4; ++j)
#pragma unroll
      for (int r = 0; r < 8; ++r) {
        const int mA = i0 + 16 * i + ((lane >> 4) << 3) + r;
        Sp[(size_t)mA * NN + j0 + 16 * j + cn] = acc[i][j][r] * scale;
      }
}

// ---------------------------------------------------------------------------
// P @ V, normalization folded into the epilogue (probs are un-normalized exp
// in bf16; rowsum holds the fp32 denominator). Wave computes 64 rows x dh=32
// (4 P-frags + 2 Vt-frags up-front, 8 wmma per k-step over K=1024).
// grid (4, 1, B*H), block 128.
// ---------------------------------------------------------------------------
__global__ void __launch_bounds__(128, 1)
attn_pv(const __bf16* __restrict__ P, const __bf16* __restrict__ Vt,
        const float* __restrict__ rowsum, float* __restrict__ Of,
        __bf16* __restrict__ Ob) {
  const int wave = threadIdx.x >> 5, lane = threadIdx.x & 31;
  const int bh = blockIdx.z, b = bh >> 3, h = bh & 7;
  const int i0 = (blockIdx.x * 4 + wave) * 64;
  const int kb = (lane < 16) ? 0 : 8;
  const __bf16* Ap0 = P + (size_t)(bh * NN + i0 + (lane & 15)) * NN + kb;
  const __bf16* Ap1 = Ap0 + (size_t)16 * NN;
  const __bf16* Ap2 = Ap0 + (size_t)32 * NN;
  const __bf16* Ap3 = Ap0 + (size_t)48 * NN;
  const __bf16* Bp0 = Vt + (size_t)(bh * DH + (lane & 15)) * NN + kb;
  const __bf16* Bp1 = Bp0 + (size_t)16 * NN;
  v8f acc00 = {}, acc01 = {}, acc10 = {}, acc11 = {};
  v8f acc20 = {}, acc21 = {}, acc30 = {}, acc31 = {};
  for (int k0 = 0; k0 < NN; k0 += 32) {
    v16bf b0 = load_frag(Bp0 + k0);
    v16bf b1 = load_frag(Bp1 + k0);
    v16bf a0 = load_frag(Ap0 + k0);
    v16bf a1 = load_frag(Ap1 + k0);
    v16bf a2 = load_frag(Ap2 + k0);
    v16bf a3 = load_frag(Ap3 + k0);
    acc00 = wmma_bf16(a0, b0, acc00);
    acc01 = wmma_bf16(a0, b1, acc01);
    acc10 = wmma_bf16(a1, b0, acc10);
    acc11 = wmma_bf16(a1, b1, acc11);
    acc20 = wmma_bf16(a2, b0, acc20);
    acc21 = wmma_bf16(a2, b1, acc21);
    acc30 = wmma_bf16(a3, b0, acc30);
    acc31 = wmma_bf16(a3, b1, acc31);
  }
  const int cn = lane & 15;
  v8f a0s[4] = {acc00, acc10, acc20, acc30};
  v8f a1s[4] = {acc01, acc11, acc21, acc31};
#pragma unroll
  for (int i = 0; i < 4; ++i)
#pragma unroll
    for (int r = 0; r < 8; ++r) {
      const int n = i0 + 16 * i + ((lane >> 4) << 3) + r;
      const float inv = 1.0f / rowsum[bh * NN + n];
      const size_t o0 = (size_t)(b * NN + n) * CC + h * DH + cn;
      const float v0 = a0s[i][r] * inv;
      const float v1 = a1s[i][r] * inv;
      Of[o0] = v0;       Ob[o0] = f2bf(v0);
      Of[o0 + 16] = v1;  Ob[o0 + 16] = f2bf(v1);
    }
}

// ---------------------------------------------------------------------------
// VALU helpers
// ---------------------------------------------------------------------------
__global__ void layernorm_rows(const float* __restrict__ X, int ldx,
                               const float* __restrict__ g,
                               const float* __restrict__ beta,
                               float* __restrict__ Yf, int ldyf,
                               __bf16* __restrict__ Yb, int ldyb, int width,
                               int do_gelu) {
  const int row = blockIdx.x;
  const float* xr = X + (size_t)row * ldx;
  __shared__ float red[256];
  float s = 0.f;
  for (int j = threadIdx.x; j < width; j += 256) s += xr[j];
  red[threadIdx.x] = s; __syncthreads();
  for (int o = 128; o > 0; o >>= 1) {
    if (threadIdx.x < o) red[threadIdx.x] += red[threadIdx.x + o];
    __syncthreads();
  }
  const float mean = red[0] / width; __syncthreads();
  float v = 0.f;
  for (int j = threadIdx.x; j < width; j += 256) {
    const float d = xr[j] - mean; v += d * d;
  }
  red[threadIdx.x] = v; __syncthreads();
  for (int o = 128; o > 0; o >>= 1) {
    if (threadIdx.x < o) red[threadIdx.x] += red[threadIdx.x + o];
    __syncthreads();
  }
  const float inv = rsqrtf(red[0] / width + 1e-5f);
  for (int j = threadIdx.x; j < width; j += 256) {
    float val = (xr[j] - mean) * inv * g[j] + beta[j];
    if (do_gelu) val = 0.5f * val * (1.0f + erff(val * 0.70710678118f));
    if (Yf) Yf[(size_t)row * ldyf + j] = val;
    if (Yb) Yb[(size_t)row * ldyb + j] = f2bf(val);
  }
}

__global__ void softmax_rows(const float* __restrict__ S,
                             __bf16* __restrict__ P,
                             float* __restrict__ rowsum) {
  const int row = blockIdx.x;
  const float* s = S + (size_t)row * NN;
  __bf16* p = P + (size_t)row * NN;
  __shared__ float red[256];
  float mx = -3.4e38f;
  for (int j = threadIdx.x; j < NN; j += 256) mx = fmaxf(mx, s[j]);
  red[threadIdx.x] = mx; __syncthreads();
  for (int o = 128; o > 0; o >>= 1) {
    if (threadIdx.x < o)
      red[threadIdx.x] = fmaxf(red[threadIdx.x], red[threadIdx.x + o]);
    __syncthreads();
  }
  mx = red[0]; __syncthreads();
  float sum = 0.f;
  for (int j = threadIdx.x; j < NN; j += 256) {
    const float e = __expf(s[j] - mx);
    p[j] = f2bf(e);
    sum += e;
  }
  red[threadIdx.x] = sum; __syncthreads();
  for (int o = 128; o > 0; o >>= 1) {
    if (threadIdx.x < o) red[threadIdx.x] += red[threadIdx.x + o];
    __syncthreads();
  }
  if (threadIdx.x == 0) rowsum[row] = red[0];
}

// Vt[bh][d][n] = qkv_bf[b*N + n][512 + h*32 + d]
__global__ void v_transpose(const __bf16* __restrict__ qkvb,
                            __bf16* __restrict__ Vt) {
  const int idx = blockIdx.x * 256 + threadIdx.x;   // B*H*32*1024 total
  const int n = idx & (NN - 1);
  const int rest = idx >> 10;
  const int d = rest & 31;
  const int bh = rest >> 5;
  const int b = bh >> 3, h = bh & 7;
  Vt[idx] = qkvb[(size_t)(b * NN + n) * 768 + 2 * CC + h * DH + d];
}

// ibpp: silu(groups-of-8 @ Win + bin) @ Wout + bout  -> catb cols [256,512)
__global__ void ibpp_kernel(const float* __restrict__ h1f,
                            const float* __restrict__ Win,
                            const float* __restrict__ bin,
                            const float* __restrict__ Wout,
                            const float* __restrict__ bout,
                            __bf16* __restrict__ catb) {
  const int idx = blockIdx.x * 256 + threadIdx.x;   // MM*32 items
  const int row = idx >> 5;
  const int gi = idx & 31;
  const float* v = h1f + (size_t)row * CC + gi * 8;
  float t[8], u[8];
#pragma unroll
  for (int i = 0; i < 8; ++i) t[i] = bin[i];
#pragma unroll
  for (int j = 0; j < 8; ++j)
#pragma unroll
    for (int i = 0; i < 8; ++i) t[i] += v[j] * Win[j * 8 + i];
#pragma unroll
  for (int i = 0; i < 8; ++i) t[i] = t[i] / (1.0f + __expf(-t[i])); // silu
#pragma unroll
  for (int i = 0; i < 8; ++i) u[i] = bout[i];
#pragma unroll
  for (int j = 0; j < 8; ++j)
#pragma unroll
    for (int i = 0; i < 8; ++i) u[i] += t[j] * Wout[j * 8 + i];
#pragma unroll
  for (int i = 0; i < 8; ++i)
    catb[(size_t)row * 512 + 256 + gi * 8 + i] = f2bf(u[i]);
}

// Direct small conv (offset convs, OC = 18 or 50). Input channel-last.
__global__ void conv_direct(const float* __restrict__ xin, int ldx, int xoff,
                            const float* __restrict__ w,
                            const float* __restrict__ bias,
                            float* __restrict__ out, int ldo, int OC, int Kk,
                            int pad) {
  const int pr = blockIdx.x;                 // b*1024 + y*32 + x
  const int b = pr >> 10, pix = pr & 1023, y = pix >> 5, x = pix & 31;
  for (int oc = threadIdx.x; oc < OC; oc += blockDim.x) {
    float acc = bias[oc];
    for (int ky = 0; ky < Kk; ++ky) {
      const int yy = y + ky - pad;
      if (yy < 0 || yy >= HH) continue;
      for (int kx = 0; kx < Kk; ++kx) {
        const int xx = x + kx - pad;
        if (xx < 0 || xx >= WW) continue;
        const float* xr = xin + (size_t)(b * NN + yy * WW + xx) * ldx + xoff;
        const float* wr = w + ((size_t)oc * H2C * Kk + ky) * Kk + kx;
        for (int c = 0; c < H2C; ++c) acc += xr[c] * wr[(size_t)c * Kk * Kk];
      }
    }
    out[(size_t)pr * ldo + oc] = acc;
  }
}

// Bilinear deform sampling -> bf16 im2col (col index = c*K2 + t)
__global__ void deform_gather(const float* __restrict__ xin, int ldx, int xoff,
                              const float* __restrict__ off, int ldo,
                              __bf16* __restrict__ Scol, int K, int pad) {
  const int pr = blockIdx.x;
  const int b = pr >> 10, pix = pr & 1023, y = pix >> 5, x = pix & 31;
  const int K2 = K * K;
  for (int t = 0; t < K2; ++t) {
    const float py = (float)(y - pad + t / K) + off[(size_t)pr * ldo + 2 * t];
    const float px = (float)(x - pad + t % K) + off[(size_t)pr * ldo + 2 * t + 1];
    const float y0f = floorf(py), x0f = floorf(px);
    const float dy = py - y0f, dx = px - x0f;
    const int y0 = (int)y0f, x0 = (int)x0f;
    float wgt[4]; int ry[4], rx[4];
    const float wys[2] = {1.0f - dy, dy};
    const float wxs[2] = {1.0f - dx, dx};
#pragma unroll
    for (int a = 0; a < 4; ++a) {
      const int iy = y0 + (a >> 1), ix = x0 + (a & 1);
      const bool valid = (iy >= 0) && (iy <= HH - 1) && (ix >= 0) && (ix <= WW - 1);
      ry[a] = min(max(iy, 0), HH - 1);
      rx[a] = min(max(ix, 0), WW - 1);
      wgt[a] = valid ? wys[a >> 1] * wxs[a & 1] : 0.0f;
    }
    for (int c = threadIdx.x; c < H2C; c += blockDim.x) {
      float s = 0.f;
#pragma unroll
      for (int a = 0; a < 4; ++a)
        s += wgt[a] * xin[(size_t)(b * NN + ry[a] * WW + rx[a]) * ldx + xoff + c];
      Scol[((size_t)pr * H2C + c) * K2 + t] = f2bf(s);
    }
  }
}

// Zero-padded im2col of channel-last dcv -> bf16 (for the second plain conv)
__global__ void im2col_pad(const float* __restrict__ dcv,
                           __bf16* __restrict__ Dcol, int K, int pad) {
  const int pr = blockIdx.x;
  const int b = pr >> 10, pix = pr & 1023, y = pix >> 5, x = pix & 31;
  const int K2 = K * K;
  for (int t = 0; t < K2; ++t) {
    const int yy = y + t / K - pad, xx = x + t % K - pad;
    const bool inb = (yy >= 0) && (yy < HH) && (xx >= 0) && (xx < WW);
    const float* src = inb ? dcv + (size_t)(b * NN + yy * WW + xx) * H2C : nullptr;
    for (int c = threadIdx.x; c < H2C; c += blockDim.x)
      Dcol[((size_t)pr * H2C + c) * K2 + t] = f2bf(inb ? src[c] : 0.0f);
  }
}

__global__ void convert_bf16(const float* __restrict__ in,
                             __bf16* __restrict__ out, int n) {
  const int i = blockIdx.x * 256 + threadIdx.x;
  if (i < n) out[i] = f2bf(in[i]);
}

// out[c][r] = bf16(in[r][c])  (weights to [N x K] for B-fragment loads)
__global__ void transpose_convert(const float* __restrict__ in,
                                  __bf16* __restrict__ out, int R, int Ccols) {
  const int i = blockIdx.x * 256 + threadIdx.x;
  if (i >= R * Ccols) return;
  const int r = i / Ccols, c = i % Ccols;
  out[(size_t)c * R + r] = f2bf(in[i]);
}

// ---------------------------------------------------------------------------
extern "C" void kernel_launch(void* const* d_in, const int* in_sizes, int n_in,
                              void* d_out, int out_size, void* d_ws,
                              size_t ws_size, hipStream_t stream) {
  const float* x        = (const float*)d_in[0];
  const float* ln1_g    = (const float*)d_in[1];
  const float* ln1_b    = (const float*)d_in[2];
  const float* qkv_w    = (const float*)d_in[3];
  const float* qkv_b    = (const float*)d_in[4];
  const float* merge_w  = (const float*)d_in[5];
  const float* merge_b  = (const float*)d_in[6];
  const float* ibpp_in_w  = (const float*)d_in[7];
  const float* ibpp_in_b  = (const float*)d_in[8];
  const float* ibpp_out_w = (const float*)d_in[9];
  const float* ibpp_out_b = (const float*)d_in[10];
  const float* proj_w   = (const float*)d_in[11];
  const float* proj_b   = (const float*)d_in[12];
  const float* ln2_g    = (const float*)d_in[13];
  const float* ln2_b    = (const float*)d_in[14];
  const float* fc1_w    = (const float*)d_in[15];
  const float* fc1_b    = (const float*)d_in[16];
  const float* dc3_off_w = (const float*)d_in[17];
  const float* dc3_off_b = (const float*)d_in[18];
  const float* dc3_w    = (const float*)d_in[19];
  const float* dc3_b    = (const float*)d_in[20];
  const float* dc5_off_w = (const float*)d_in[21];
  const float* dc5_off_b = (const float*)d_in[22];
  const float* dc5_w    = (const float*)d_in[23];
  const float* dc5_b    = (const float*)d_in[24];
  const float* ffn_ln_g = (const float*)d_in[25];
  const float* ffn_ln_b = (const float*)d_in[26];
  const float* fc2_w    = (const float*)d_in[27];
  const float* fc2_b    = (const float*)d_in[28];
  float* out = (float*)d_out;

  // workspace carve-out (256B aligned slabs)
  size_t off = 0;
  auto alloc = [&](size_t bytes) -> void* {
    void* p = (char*)d_ws + off;
    off += (bytes + 255) & ~(size_t)255;
    return p;
  };
  float*  h1f    = (float*)alloc((size_t)MM * CC * 4);
  __bf16* h1b    = (__bf16*)alloc((size_t)MM * CC * 2);
  __bf16* qkvb   = (__bf16*)alloc((size_t)MM * 768 * 2);
  __bf16* qkv_wt = (__bf16*)alloc((size_t)768 * CC * 2);
  __bf16* merge_wt = (__bf16*)alloc((size_t)CC * CC * 2);
  __bf16* proj_wt  = (__bf16*)alloc((size_t)CC * 512 * 2);
  __bf16* fc1_wt   = (__bf16*)alloc((size_t)HID * CC * 2);
  __bf16* fc2_wt   = (__bf16*)alloc((size_t)CC * HID * 2);
  __bf16* dc3_wb   = (__bf16*)alloc((size_t)H2C * (H2C * 9) * 2);
  __bf16* dc5_wb   = (__bf16*)alloc((size_t)H2C * (H2C * 25) * 2);
  float*  scores = (float*)alloc((size_t)BB * HEADS * NN * NN * 4);
  __bf16* probs  = (__bf16*)alloc((size_t)BB * HEADS * NN * NN * 2);
  float*  rowsum = (float*)alloc((size_t)BB * HEADS * NN * 4);
  __bf16* vT     = (__bf16*)alloc((size_t)BB * HEADS * DH * NN * 2);
  float*  attnf  = (float*)alloc((size_t)MM * CC * 4);
  __bf16* attnb  = (__bf16*)alloc((size_t)MM * CC * 2);
  __bf16* catb   = (__bf16*)alloc((size_t)MM * 512 * 2);
  float*  xa     = (float*)alloc((size_t)MM * CC * 4);
  __bf16* h2b    = (__bf16*)alloc((size_t)MM * CC * 2);
  float*  hfc1   = (float*)alloc((size_t)MM * HID * 4);
  float*  offbuf = (float*)alloc((size_t)MM * 64 * 4);
  __bf16* scol   = (__bf16*)alloc((size_t)MM * (H2C * 25) * 2);
  float*  dcv    = (float*)alloc((size_t)MM * H2C * 4);
  float*  hcat   = (float*)alloc((size_t)MM * HID * 4);
  __bf16* ffnb   = (__bf16*)alloc((size_t)MM * HID * 2);

  const dim3 blk256(256), blk128(128), blk64(64);
  auto cdiv = [](int a, int b) { return (a + b - 1) / b; };

  // -- weight prep (bf16, B-operand transposed to [N x K]) --
  transpose_convert<<<cdiv(CC * 768, 256), blk256, 0, stream>>>(qkv_w, qkv_wt, CC, 768);
  transpose_convert<<<cdiv(CC * CC, 256), blk256, 0, stream>>>(merge_w, merge_wt, CC, CC);
  transpose_convert<<<cdiv(512 * CC, 256), blk256, 0, stream>>>(proj_w, proj_wt, 512, CC);
  transpose_convert<<<cdiv(CC * HID, 256), blk256, 0, stream>>>(fc1_w, fc1_wt, CC, HID);
  transpose_convert<<<cdiv(HID * CC, 256), blk256, 0, stream>>>(fc2_w, fc2_wt, HID, CC);
  convert_bf16<<<cdiv(H2C * H2C * 9, 256), blk256, 0, stream>>>(dc3_w, dc3_wb, H2C * H2C * 9);
  convert_bf16<<<cdiv(H2C * H2C * 25, 256), blk256, 0, stream>>>(dc5_w, dc5_wb, H2C * H2C * 25);

  // -- attention branch --
  layernorm_rows<<<MM, blk256, 0, stream>>>(x, CC, ln1_g, ln1_b, h1f, CC, h1b, CC, CC, 0);
  gemm_bf16_wmma<<<dim3(MM / 64, 768 / 128), blk128, 0, stream>>>(
      h1b, CC, qkv_wt, CC, qkv_b, nullptr, 0, nullptr, 0, qkvb, 768, CC);
  v_transpose<<<(BB * HEADS * DH * NN) / 256, blk256, 0, stream>>>(qkvb, vT);
  attn_scores<<<dim3(NN / 32, NN / 256, BB * HEADS), blk128, 0, stream>>>(
      qkvb, scores, 0.17677669529663687f);  // (dh)^-0.5
  softmax_rows<<<BB * HEADS * NN, blk256, 0, stream>>>(scores, probs, rowsum);
  attn_pv<<<dim3(NN / 256, 1, BB * HEADS), blk128, 0, stream>>>(probs, vT, rowsum, attnf, attnb);
  gemm_bf16_wmma<<<dim3(MM / 64, CC / 128), blk128, 0, stream>>>(
      attnb, CC, merge_wt, CC, merge_b, nullptr, 0, nullptr, 0, catb, 512, CC);
  ibpp_kernel<<<(MM * 32) / 256, blk256, 0, stream>>>(
      h1f, ibpp_in_w, ibpp_in_b, ibpp_out_w, ibpp_out_b, catb);
  gemm_bf16_wmma<<<dim3(MM / 64, CC / 128), blk128, 0, stream>>>(
      catb, 512, proj_wt, 512, proj_b, x, CC, xa, CC, nullptr, 0, 512);

  // -- EHFF branch --
  layernorm_rows<<<MM, blk256, 0, stream>>>(xa, CC, ln2_g, ln2_b, nullptr, 0, h2b, CC, CC, 0);
  gemm_bf16_wmma<<<dim3(MM / 64, HID / 128), blk128, 0, stream>>>(
      h2b, CC, fc1_wt, CC, fc1_b, nullptr, 0, hfc1, HID, nullptr, 0, CC);

  // branch 1: 3x3 deform conv + plain conv (K = C*9 = 4608)
  conv_direct<<<MM, blk64, 0, stream>>>(hfc1, HID, 0, dc3_off_w, dc3_off_b,
                                        offbuf, 18, 18, 3, 1);
  deform_gather<<<MM, blk128, 0, stream>>>(hfc1, HID, 0, offbuf, 18, scol, 3, 1);
  gemm_bf16_wmma<<<dim3(MM / 64, H2C / 128), blk128, 0, stream>>>(
      scol, H2C * 9, dc3_wb, H2C * 9, dc3_b, nullptr, 0, dcv, H2C, nullptr, 0,
      H2C * 9);
  im2col_pad<<<MM, blk128, 0, stream>>>(dcv, scol, 3, 1);
  gemm_bf16_wmma<<<dim3(MM / 64, H2C / 128), blk128, 0, stream>>>(
      scol, H2C * 9, dc3_wb, H2C * 9, dc3_b, nullptr, 0, hcat, HID, nullptr, 0,
      H2C * 9);

  // branch 2: 5x5 deform conv + plain conv (K = C*25 = 12800)
  conv_direct<<<MM, blk64, 0, stream>>>(hfc1, HID, H2C, dc5_off_w, dc5_off_b,
                                        offbuf, 50, 50, 5, 2);
  deform_gather<<<MM, blk128, 0, stream>>>(hfc1, HID, H2C, offbuf, 50, scol, 5, 2);
  gemm_bf16_wmma<<<dim3(MM / 64, H2C / 128), blk128, 0, stream>>>(
      scol, H2C * 25, dc5_wb, H2C * 25, dc5_b, nullptr, 0, dcv, H2C, nullptr, 0,
      H2C * 25);
  im2col_pad<<<MM, blk128, 0, stream>>>(dcv, scol, 5, 2);
  gemm_bf16_wmma<<<dim3(MM / 64, H2C / 128), blk128, 0, stream>>>(
      scol, H2C * 25, dc5_wb, H2C * 25, dc5_b, nullptr, 0, hcat + H2C, HID,
      nullptr, 0, H2C * 25);

  // LN + exact GELU, then fc2 with residual into d_out
  layernorm_rows<<<MM, blk256, 0, stream>>>(hcat, HID, ffn_ln_g, ffn_ln_b,
                                            nullptr, 0, ffnb, HID, HID, 1);
  gemm_bf16_wmma<<<dim3(MM / 64, CC / 128), blk128, 0, stream>>>(
      ffnb, HID, fc2_wt, HID, fc2_b, xa, CC, out, CC, nullptr, 0, HID);

  (void)in_sizes; (void)n_in; (void)out_size; (void)ws_size;
}
